// MeshGraphNet_84576495992987
// MI455X (gfx1250) — matrix operand order repaired and checked by reading, not compiled, not run on hardware
//
#include <hip/hip_runtime.h>

#define HID 128
#define NLAYERS 4
#define USE_ASYNC_GATHER 1

typedef __attribute__((ext_vector_type(16))) __bf16 v16bf;
typedef __attribute__((ext_vector_type(8)))  __bf16 v8bf;
typedef __attribute__((ext_vector_type(8)))  float  v8f;
typedef unsigned int   uint_t;

__device__ __forceinline__ unsigned short f32_to_bf16(float f) {
  unsigned int u = __float_as_uint(f);
  u += 0x7FFFu + ((u >> 16) & 1u);            // round-to-nearest-even
  return (unsigned short)(u >> 16);
}

union ABf { v16bf v; v8bf h[2]; };

// LDS byte offset of a __shared__ object (generic -> AS3 -> 32-bit DS address)
__device__ __forceinline__ unsigned lds_addr_of(const void* p) {
  return (unsigned)(uintptr_t)(__attribute__((address_space(3))) const void*)p;
}

#if USE_ASYNC_GATHER
// CDNA5 async memory->LDS DMA (GVS mode: SGPR base + 32-bit VGPR byte offset),
// LDS dest address per lane in VDST. Tracked by ASYNCcnt.
__device__ __forceinline__ void async_ld_b128(unsigned lds_off, unsigned g_off, const void* base) {
  asm volatile("global_load_async_to_lds_b128 %0, %1, %2"
               :: "v"(lds_off), "v"(g_off), "s"((unsigned long long)(uintptr_t)base)
               : "memory");
}
__device__ __forceinline__ void wait_async0() {
  asm volatile("s_wait_asynccnt 0" ::: "memory");
}
#endif

// One fused GEMM pass: 32xK (LDS, bf16) x Kx128 (packed tiles, bf16) -> two 16x16 f32 accs per wave.
// Packed B layout: tile (kt,nt) -> 32 lanes x 16 contiguous bf16 -> one 32B load per lane per k-step.
__device__ __forceinline__ void wmma_pair(
    const unsigned short* __restrict__ ldsA, int strideA,
    const unsigned short* __restrict__ wpk, int K,
    v8f& acc0, v8f& acc1, int wv, int ln)
{
  const int col16 = ln & 15;
  const int hi    = ln >> 4;
  const int KT    = K >> 5;
  for (int kt = 0; kt < KT; ++kt) {
    v16bf b = *(const v16bf*)(wpk + (size_t)(((kt << 3) + wv) * 32 + ln) * 16);
    const int kb = (kt << 5) + (hi << 3);     // A lanes 0-15: K 0-7/16-23, lanes 16-31: K 8-15/24-31
    const unsigned short* r0 = ldsA + col16 * strideA + kb;
    const unsigned short* r1 = ldsA + (col16 + 16) * strideA + kb;
    ABf a0, a1;
    a0.h[0] = *(const v8bf*)(r0);
    a0.h[1] = *(const v8bf*)(r0 + 16);
    a1.h[0] = *(const v8bf*)(r1);
    a1.h[1] = *(const v8bf*)(r1 + 16);
    acc0 = __builtin_amdgcn_wmma_f32_16x16x32_bf16(false, a0.v, false, b, (short)0, acc0, false, false);
    acc1 = __builtin_amdgcn_wmma_f32_16x16x32_bf16(false, a1.v, false, b, (short)0, acc1, false, false);
  }
}

__device__ __forceinline__ void epi_relu_bf16(const v8f& a, unsigned short* lds, int stride,
                                              int rowbase, int col, float bias) {
#pragma unroll
  for (int r = 0; r < 8; ++r) {
    float x = a[r] + bias;
    x = x > 0.f ? x : 0.f;
    lds[(rowbase + r) * stride + col] = f32_to_bf16(x);
  }
}

__device__ __forceinline__ void epi_f32(const v8f& a, float* lds, int stride,
                                        int rowbase, int col, float bias) {
#pragma unroll
  for (int r = 0; r < 8; ++r)
    lds[(rowbase + r) * stride + col] = a[r] + bias;
}

__device__ __forceinline__ void wave_red2(float& s, float& s2) {
#pragma unroll
  for (int off = 16; off > 0; off >>= 1) {
    s  += __shfl_xor(s, off, 32);
    s2 += __shfl_xor(s2, off, 32);
  }
}

// ---------------- weight packing (f32 row-major [K,N] -> WMMA B-tile bf16) ----------------
__global__ __launch_bounds__(256) void pack_weight(const float* __restrict__ w,
                                                   unsigned short* __restrict__ dst,
                                                   int K, int N, int total) {
  int idx = blockIdx.x * 256 + threadIdx.x;
  if (idx >= total) return;
  int j    = idx & 15;
  int lane = (idx >> 4) & 31;
  int tile = idx >> 9;
  int NT   = N >> 4;
  int kt   = tile / NT;
  int nt   = tile - kt * NT;
  int k = (kt << 5) + ((lane >> 4) << 3) + (j & 7) + ((j >> 3) << 4);
  int n = (nt << 4) + (lane & 15);
  dst[idx] = f32_to_bf16(w[(size_t)k * N + n]);
}

__global__ __launch_bounds__(256) void zero_f32(float* __restrict__ p, int n) {
  int i = blockIdx.x * 256 + threadIdx.x;
  if (i < n) p[i] = 0.f;
}

// ---------------- encoders: normalize -> Linear(INF->128,VALU)+ReLU -> Linear(128,WMMA) -> LN ----------------
template <int INF>
__global__ __launch_bounds__(256) void encoder_kernel(
    const float* __restrict__ xin, const float* __restrict__ meanv, const float* __restrict__ stdv,
    const float* __restrict__ w1, const float* __restrict__ b1,
    const unsigned short* __restrict__ w2pk, const float* __restrict__ b2,
    const float* __restrict__ g, const float* __restrict__ be,
    float* __restrict__ out32, unsigned short* __restrict__ out16, int nrows)
{
  __shared__ float XS[32 * INF];
  __shared__ unsigned short H1[32 * 136];
  __shared__ float O[32 * 132];
  const int r0 = blockIdx.x * 32;
  const int tid = threadIdx.x;
  for (int t = tid; t < 32 * INF; t += 256) {
    int row = t / INF, k = t - row * INF;
    int rr = min(r0 + row, nrows - 1);
    XS[t] = (xin[(size_t)rr * INF + k] - meanv[k]) / stdv[k];
  }
  __syncthreads();
  for (int t = tid; t < 32 * HID; t += 256) {
    int row = t >> 7, n = t & 127;
    float acc = b1[n];
#pragma unroll
    for (int k = 0; k < INF; ++k) acc += XS[row * INF + k] * w1[k * HID + n];
    acc = acc > 0.f ? acc : 0.f;
    H1[row * 136 + n] = f32_to_bf16(acc);
  }
  __syncthreads();
  const int wv = tid >> 5, ln = tid & 31, col16 = ln & 15, hi = ln >> 4;
  v8f a0 = {}, a1 = {};
  wmma_pair(H1, 136, w2pk, HID, a0, a1, wv, ln);
  const float bias = b2[wv * 16 + col16];
  epi_f32(a0, O, 132, hi * 8, wv * 16 + col16, bias);
  epi_f32(a1, O, 132, 16 + hi * 8, wv * 16 + col16, bias);
  __syncthreads();
  for (int rr = 0; rr < 4; ++rr) {
    int row = wv * 4 + rr;
    int node = r0 + row;
    float vals[4], s = 0.f, s2 = 0.f;
#pragma unroll
    for (int j = 0; j < 4; ++j) { float xv = O[row * 132 + ln + 32 * j]; vals[j] = xv; s += xv; s2 += xv * xv; }
    wave_red2(s, s2);
    float mu  = s * (1.f / 128.f);
    float inv = rsqrtf(s2 * (1.f / 128.f) - mu * mu + 1e-5f);
    if (node < nrows) {
#pragma unroll
      for (int j = 0; j < 4; ++j) {
        int c = ln + 32 * j;
        float y = (vals[j] - mu) * inv * g[c] + be[c];
        out32[(size_t)node * HID + c] = y;
        out16[(size_t)node * HID + c] = f32_to_bf16(y);
      }
    }
  }
}

// ---------------- edge update: gather concat -> MLP(384->128->128) -> LN -> +e -> scatter-sum ----------------
__global__ __launch_bounds__(256) void edge_update_kernel(
    const unsigned short* __restrict__ h16,
    const int* __restrict__ srcI, const int* __restrict__ dstI,
    const unsigned short* __restrict__ w1pk, const float* __restrict__ b1,
    const unsigned short* __restrict__ w2pk, const float* __restrict__ b2,
    const float* __restrict__ g, const float* __restrict__ be,
    float* __restrict__ e32, unsigned short* __restrict__ e16,
    float* __restrict__ agg, int nE)
{
  __shared__ unsigned short A[32 * 392];   // 32 x 384 bf16 (+8 pad); row = [h[dst] | h[src] | e]
  __shared__ unsigned short H1[32 * 136];
  __shared__ float O[32 * 132];
  const int e0 = blockIdx.x * 32;
  const int tid = threadIdx.x;
#if USE_ASYNC_GATHER
  // 16 x 16B chunks per 256B source row; class-uniform passes, async DMA straight into LDS.
#pragma unroll
  for (int it = 0; it < 2; ++it) {
    int idx = it * 256 + tid;              // 0..511
    int row = idx >> 4, c = idx & 15;
    int edge = min(e0 + row, nE - 1);
    unsigned ldsrow = lds_addr_of(&A[row * 392]);
    unsigned coff = (unsigned)c * 16u;
    async_ld_b128(ldsrow + coff,        (unsigned)dstI[edge] * 256u + coff, h16);
    async_ld_b128(ldsrow + 256u + coff, (unsigned)srcI[edge] * 256u + coff, h16);
    async_ld_b128(ldsrow + 512u + coff, (unsigned)edge * 256u + coff,       e16);
  }
  wait_async0();
#else
  for (int t = tid; t < 32 * 48; t += 256) {           // 48 x 16B chunks per row
    int row = t / 48, c = t - row * 48;
    int edge = min(e0 + row, nE - 1);
    uint4 v;
    if (c < 16)      v = ((const uint4*)h16)[(size_t)dstI[edge] * 16 + c];
    else if (c < 32) v = ((const uint4*)h16)[(size_t)srcI[edge] * 16 + (c - 16)];
    else             v = ((const uint4*)e16)[(size_t)edge * 16 + (c - 32)];
    *(uint4*)&A[row * 392 + c * 8] = v;
  }
#endif
  __syncthreads();
  const int wv = tid >> 5, ln = tid & 31, col16 = ln & 15, hi = ln >> 4;
  v8f a0 = {}, a1 = {};
  wmma_pair(A, 392, w1pk, 384, a0, a1, wv, ln);
  const float bias1 = b1[wv * 16 + col16];
  epi_relu_bf16(a0, H1, 136, hi * 8, wv * 16 + col16, bias1);
  epi_relu_bf16(a1, H1, 136, 16 + hi * 8, wv * 16 + col16, bias1);
  __syncthreads();
  v8f c0 = {}, c1 = {};
  wmma_pair(H1, 136, w2pk, 128, c0, c1, wv, ln);
  const float bias2 = b2[wv * 16 + col16];
  epi_f32(c0, O, 132, hi * 8, wv * 16 + col16, bias2);
  epi_f32(c1, O, 132, 16 + hi * 8, wv * 16 + col16, bias2);
  __syncthreads();
  for (int rr = 0; rr < 4; ++rr) {
    int row = wv * 4 + rr;
    int edge = e0 + row;
    float vals[4], s = 0.f, s2 = 0.f;
#pragma unroll
    for (int j = 0; j < 4; ++j) { float xv = O[row * 132 + ln + 32 * j]; vals[j] = xv; s += xv; s2 += xv * xv; }
    wave_red2(s, s2);
    float mu  = s * (1.f / 128.f);
    float inv = rsqrtf(s2 * (1.f / 128.f) - mu * mu + 1e-5f);
    if (edge < nE) {
      int d = dstI[edge];
#pragma unroll
      for (int j = 0; j < 4; ++j) {
        int c = ln + 32 * j;
        float y  = (vals[j] - mu) * inv * g[c] + be[c];
        float en = y + e32[(size_t)edge * HID + c];
        e32[(size_t)edge * HID + c] = en;
        e16[(size_t)edge * HID + c] = f32_to_bf16(en);
        atomicAdd(agg + (size_t)d * HID + c, en);      // segment_sum over dst
      }
    }
  }
}

// ---------------- node update: concat[h,agg] -> MLP(256->128->128) -> LN -> h += ----------------
__global__ __launch_bounds__(256) void node_update_kernel(
    const unsigned short* __restrict__ h16in, const float* __restrict__ agg,
    const unsigned short* __restrict__ w1pk, const float* __restrict__ b1,
    const unsigned short* __restrict__ w2pk, const float* __restrict__ b2,
    const float* __restrict__ g, const float* __restrict__ be,
    float* __restrict__ h32, unsigned short* __restrict__ h16out, int nN)
{
  __shared__ unsigned short A[32 * 264];   // 32 x 256 bf16 (+8 pad); row = [h | bf16(agg)]
  __shared__ unsigned short H1[32 * 136];
  __shared__ float O[32 * 132];
  const int n0 = blockIdx.x * 32;
  const int tid = threadIdx.x;
#if USE_ASYNC_GATHER
#pragma unroll
  for (int it = 0; it < 2; ++it) {
    int idx = it * 256 + tid;
    int row = idx >> 4, c = idx & 15;
    int node = min(n0 + row, nN - 1);
    unsigned coff = (unsigned)c * 16u;
    async_ld_b128(lds_addr_of(&A[row * 264]) + coff, (unsigned)node * 256u + coff, h16in);
  }
#else
  for (int t = tid; t < 32 * 16; t += 256) {
    int row = t >> 4, c = t & 15;
    int node = min(n0 + row, nN - 1);
    *(uint4*)&A[row * 264 + c * 8] = ((const uint4*)h16in)[(size_t)node * 16 + c];
  }
#endif
  for (int t = tid; t < 32 * 64; t += 256) {           // agg half: cvt f32->bf16, packed dwords
    int row = t >> 6, c2 = t & 63;
    int node = min(n0 + row, nN - 1);
    float f0 = agg[(size_t)node * HID + c2 * 2];
    float f1 = agg[(size_t)node * HID + c2 * 2 + 1];
    uint_t v = (uint_t)f32_to_bf16(f0) | ((uint_t)f32_to_bf16(f1) << 16);
    *(uint_t*)&A[row * 264 + 128 + c2 * 2] = v;
  }
#if USE_ASYNC_GATHER
  wait_async0();
#endif
  __syncthreads();
  const int wv = tid >> 5, ln = tid & 31, col16 = ln & 15, hi = ln >> 4;
  v8f a0 = {}, a1 = {};
  wmma_pair(A, 264, w1pk, 256, a0, a1, wv, ln);
  const float bias1 = b1[wv * 16 + col16];
  epi_relu_bf16(a0, H1, 136, hi * 8, wv * 16 + col16, bias1);
  epi_relu_bf16(a1, H1, 136, 16 + hi * 8, wv * 16 + col16, bias1);
  __syncthreads();
  v8f c0 = {}, c1 = {};
  wmma_pair(H1, 136, w2pk, 128, c0, c1, wv, ln);
  const float bias2 = b2[wv * 16 + col16];
  epi_f32(c0, O, 132, hi * 8, wv * 16 + col16, bias2);
  epi_f32(c1, O, 132, 16 + hi * 8, wv * 16 + col16, bias2);
  __syncthreads();
  for (int rr = 0; rr < 4; ++rr) {
    int row = wv * 4 + rr;
    int node = n0 + row;
    float vals[4], s = 0.f, s2 = 0.f;
#pragma unroll
    for (int j = 0; j < 4; ++j) { float xv = O[row * 132 + ln + 32 * j]; vals[j] = xv; s += xv; s2 += xv * xv; }
    wave_red2(s, s2);
    float mu  = s * (1.f / 128.f);
    float inv = rsqrtf(s2 * (1.f / 128.f) - mu * mu + 1e-5f);
    if (node < nN) {
#pragma unroll
      for (int j = 0; j < 4; ++j) {
        int c = ln + 32 * j;
        float y  = (vals[j] - mu) * inv * g[c] + be[c];
        float hn = h32[(size_t)node * HID + c] + y;
        h32[(size_t)node * HID + c] = hn;
        h16out[(size_t)node * HID + c] = f32_to_bf16(hn);
      }
    }
  }
}

// ---------------- decoder: relu(h@W1+b1) @ W2 + b2 ----------------
__global__ __launch_bounds__(256) void decoder_kernel(
    const unsigned short* __restrict__ h16,
    const unsigned short* __restrict__ w1pk, const float* __restrict__ b1,
    const float* __restrict__ w2, const float* __restrict__ b2,
    float* __restrict__ out, int nN)
{
  __shared__ unsigned short A[32 * 136];
  __shared__ float O[32 * 132];
  const int n0 = blockIdx.x * 32;
  const int tid = threadIdx.x;
#if USE_ASYNC_GATHER
#pragma unroll
  for (int it = 0; it < 2; ++it) {
    int idx = it * 256 + tid;
    int row = idx >> 4, c = idx & 15;
    int node = min(n0 + row, nN - 1);
    unsigned coff = (unsigned)c * 16u;
    async_ld_b128(lds_addr_of(&A[row * 136]) + coff, (unsigned)node * 256u + coff, h16);
  }
  wait_async0();
#else
  for (int t = tid; t < 32 * 16; t += 256) {
    int row = t >> 4, c = t & 15;
    int node = min(n0 + row, nN - 1);
    *(uint4*)&A[row * 136 + c * 8] = ((const uint4*)h16)[(size_t)node * 16 + c];
  }
#endif
  __syncthreads();
  const int wv = tid >> 5, ln = tid & 31, col16 = ln & 15, hi = ln >> 4;
  v8f a0 = {}, a1 = {};
  wmma_pair(A, 136, w1pk, 128, a0, a1, wv, ln);
  const float bias = b1[wv * 16 + col16];
#pragma unroll
  for (int r = 0; r < 8; ++r) {
    float x0 = a0[r] + bias; x0 = x0 > 0.f ? x0 : 0.f;
    float x1 = a1[r] + bias; x1 = x1 > 0.f ? x1 : 0.f;
    O[(hi * 8 + r) * 132 + wv * 16 + col16]      = x0;
    O[(16 + hi * 8 + r) * 132 + wv * 16 + col16] = x1;
  }
  __syncthreads();
  for (int rr = 0; rr < 4; ++rr) {
    int row = wv * 4 + rr;
    int node = n0 + row;
    float s = 0.f;
#pragma unroll
    for (int j = 0; j < 4; ++j) { int c = ln + 32 * j; s += O[row * 132 + c] * w2[c]; }
#pragma unroll
    for (int off = 16; off > 0; off >>= 1) s += __shfl_xor(s, off, 32);
    if (ln == 0 && node < nN) out[node] = s + b2[0];
  }
}

extern "C" void kernel_launch(void* const* d_in, const int* in_sizes, int n_in,
                              void* d_out, int out_size, void* d_ws, size_t ws_size,
                              hipStream_t stream)
{
  const float* x          = (const float*)d_in[0];
  const float* edge_attr  = (const float*)d_in[1];
  const int*   edge_index = (const int*)d_in[2];
  const float* mean_x     = (const float*)d_in[3];
  const float* std_x      = (const float*)d_in[4];
  const float* mean_e     = (const float*)d_in[5];
  const float* std_e      = (const float*)d_in[6];
  const float* ne_w1 = (const float*)d_in[7];
  const float* ne_b1 = (const float*)d_in[8];
  const float* ne_w2 = (const float*)d_in[9];
  const float* ne_b2 = (const float*)d_in[10];
  const float* ne_g  = (const float*)d_in[11];
  const float* ne_be = (const float*)d_in[12];
  const float* ee_w1 = (const float*)d_in[13];
  const float* ee_b1 = (const float*)d_in[14];
  const float* ee_w2 = (const float*)d_in[15];
  const float* ee_b2 = (const float*)d_in[16];
  const float* ee_g  = (const float*)d_in[17];
  const float* ee_be = (const float*)d_in[18];
  const float* pe_w1 = (const float*)d_in[19];
  const float* pe_b1 = (const float*)d_in[20];
  const float* pe_w2 = (const float*)d_in[21];
  const float* pe_b2 = (const float*)d_in[22];
  const float* pe_g  = (const float*)d_in[23];
  const float* pe_be = (const float*)d_in[24];
  const float* pn_w1 = (const float*)d_in[25];
  const float* pn_b1 = (const float*)d_in[26];
  const float* pn_w2 = (const float*)d_in[27];
  const float* pn_b2 = (const float*)d_in[28];
  const float* pn_g  = (const float*)d_in[29];
  const float* pn_be = (const float*)d_in[30];
  const float* dec_w1 = (const float*)d_in[31];
  const float* dec_b1 = (const float*)d_in[32];
  const float* dec_w2 = (const float*)d_in[33];
  const float* dec_b2 = (const float*)d_in[34];
  (void)n_in; (void)out_size; (void)ws_size;

  const int N = in_sizes[0] / 11;
  const int E = in_sizes[1] / 3;
  const int* srcI = edge_index;      // edge_index[0]
  const int* dstI = edge_index + E;  // edge_index[1]

  // ---- workspace carve-up ----
  auto up = [](size_t v) { return (v + 255) & ~(size_t)255; };
  char* ws = (char*)d_ws;
  size_t off = 0;
  unsigned short* Wpk = (unsigned short*)(ws + off); off = up(off + (size_t)507904 * 2);
  float*          h32 = (float*)(ws + off);          off = up(off + (size_t)N * HID * 4);
  unsigned short* h16 = (unsigned short*)(ws + off); off = up(off + (size_t)N * HID * 2);
  float*          aggp = (float*)(ws + off);         off = up(off + (size_t)N * HID * 4);
  float*          e32 = (float*)(ws + off);          off = up(off + (size_t)E * HID * 4);
  unsigned short* e16 = (unsigned short*)(ws + off); off = up(off + (size_t)E * HID * 2);

  unsigned short* W_ne2  = Wpk;
  unsigned short* W_ee2  = Wpk + 16384;
  unsigned short* W_dec1 = Wpk + 32768;
  unsigned short* W_pe1  = Wpk + 49152;   // + l*49152
  unsigned short* W_pe2  = Wpk + 245760;  // + l*16384
  unsigned short* W_pn1  = Wpk + 311296;  // + l*32768
  unsigned short* W_pn2  = Wpk + 442368;  // + l*16384

  dim3 blk(256);
  auto gp = [](int total) { return dim3((unsigned)((total + 255) / 256)); };

  // pack weights into WMMA B-tile bf16 layout
  pack_weight<<<gp(128 * 128), blk, 0, stream>>>(ne_w2, W_ne2, 128, 128, 128 * 128);
  pack_weight<<<gp(128 * 128), blk, 0, stream>>>(ee_w2, W_ee2, 128, 128, 128 * 128);
  pack_weight<<<gp(128 * 128), blk, 0, stream>>>(dec_w1, W_dec1, 128, 128, 128 * 128);
  for (int l = 0; l < NLAYERS; ++l) {
    pack_weight<<<gp(384 * 128), blk, 0, stream>>>(pe_w1 + (size_t)l * 384 * 128, W_pe1 + l * 49152, 384, 128, 384 * 128);
    pack_weight<<<gp(128 * 128), blk, 0, stream>>>(pe_w2 + (size_t)l * 128 * 128, W_pe2 + l * 16384, 128, 128, 128 * 128);
    pack_weight<<<gp(256 * 128), blk, 0, stream>>>(pn_w1 + (size_t)l * 256 * 128, W_pn1 + l * 32768, 256, 128, 256 * 128);
    pack_weight<<<gp(128 * 128), blk, 0, stream>>>(pn_w2 + (size_t)l * 128 * 128, W_pn2 + l * 16384, 128, 128, 128 * 128);
  }

  const int nblkN = (N + 31) / 32;
  const int nblkE = (E + 31) / 32;

  encoder_kernel<11><<<nblkN, blk, 0, stream>>>(x, mean_x, std_x, ne_w1, ne_b1, W_ne2, ne_b2, ne_g, ne_be, h32, h16, N);
  encoder_kernel<3><<<nblkE, blk, 0, stream>>>(edge_attr, mean_e, std_e, ee_w1, ee_b1, W_ee2, ee_b2, ee_g, ee_be, e32, e16, E);

  for (int l = 0; l < NLAYERS; ++l) {
    zero_f32<<<gp(N * HID), blk, 0, stream>>>(aggp, N * HID);
    edge_update_kernel<<<nblkE, blk, 0, stream>>>(h16, srcI, dstI,
        W_pe1 + l * 49152, pe_b1 + l * HID, W_pe2 + l * 16384, pe_b2 + l * HID,
        pe_g + l * HID, pe_be + l * HID, e32, e16, aggp, E);
    node_update_kernel<<<nblkN, blk, 0, stream>>>(h16, aggp,
        W_pn1 + l * 32768, pn_b1 + l * HID, W_pn2 + l * 16384, pn_b2 + l * HID,
        pn_g + l * HID, pn_be + l * HID, h32, h16, N);
  }

  decoder_kernel<<<nblkN, blk, 0, stream>>>(h16, W_dec1, dec_b1, dec_w2, dec_b2, (float*)d_out, N);
}
